// MiniS4D_16363825398537
// MI455X (gfx1250) — compile-verified, matrix operand order checked by reading, not run on hardware
//
#include <hip/hip_runtime.h>
#include <math.h>

typedef _Float16 v16h __attribute__((ext_vector_type(16)));
typedef float    v8f  __attribute__((ext_vector_type(8)));
typedef unsigned int v4u __attribute__((ext_vector_type(4)));
typedef unsigned int v8u __attribute__((ext_vector_type(8)));

#define D_MODEL 512
#define D_STATE 64
#define BATCH   16
#define SEQL    4096

// ---------------------------------------------------------------------------
// K1: build S4D conv kernel K[d, l] (f16), one block per channel d.
// K[d,l] = Re( sum_n (C*B_bar)[d,n] * exp(dtA[d,n] * l) )
// ---------------------------------------------------------------------------
__global__ void s4d_build_kernel(const float* __restrict__ log_dt,
                                 const float* __restrict__ log_A_real,
                                 const float* __restrict__ A_imag,
                                 const float* __restrict__ B_re,
                                 const float* __restrict__ B_im,
                                 const float* __restrict__ C_re,
                                 const float* __restrict__ C_im,
                                 _Float16* __restrict__ Kf16) {
  const int d = blockIdx.x;
  const int tid = threadIdx.x;
  __shared__ float sAr[D_STATE], sAi[D_STATE], sCBr[D_STATE], sCBi[D_STATE];

  if (tid < D_STATE) {
    const int idx = d * D_STATE + tid;
    float dt   = expf(log_dt[d]);
    float Ar   = -expf(log_A_real[idx]);
    float Ai   = A_imag[idx];
    float dtAr = Ar * dt, dtAi = Ai * dt;
    float e = expf(dtAr);
    float sn, cs; sincosf(dtAi, &sn, &cs);
    float Abr = e * cs, Abi = e * sn;          // A_bar = exp(dtA)
    float mr = Abr - 1.0f, mi = Abi;           // (A_bar - 1) / A
    float inv = 1.0f / (Ar * Ar + Ai * Ai);
    float qr = (mr * Ar + mi * Ai) * inv;
    float qi = (mi * Ar - mr * Ai) * inv;
    float Br = B_re[idx], Bi = B_im[idx];      // * B
    float Bbr = qr * Br - qi * Bi, Bbi = qr * Bi + qi * Br;
    float Cr = C_re[idx], Ci = C_im[idx];      // * C
    sCBr[tid] = Cr * Bbr - Ci * Bbi;
    sCBi[tid] = Cr * Bbi + Ci * Bbr;
    sAr[tid] = dtAr; sAi[tid] = dtAi;
  }
  __syncthreads();

  for (int l = tid; l < SEQL; l += blockDim.x) {
    const float fl = (float)l;
    float acc = 0.0f;
    #pragma unroll 8
    for (int n = 0; n < D_STATE; ++n) {
      float e = expf(sAr[n] * fl);
      float sn, cs; sincosf(sAi[n] * fl, &sn, &cs);
      acc += e * (sCBr[n] * cs - sCBi[n] * sn);
    }
    Kf16[d * SEQL + l] = (_Float16)acc;
  }
}

// ---------------------------------------------------------------------------
// K1b: transpose+convert u (fp32, (b,d,l)) -> uh (f16, (d,b,l)) so each
// channel's 16x4096 panel is one contiguous 128KB block for the TDM.
// ---------------------------------------------------------------------------
__global__ void s4d_u_to_f16(const float* __restrict__ u,
                             _Float16* __restrict__ uh) {
  const size_t total = (size_t)BATCH * D_MODEL * SEQL;
  const size_t step = (size_t)gridDim.x * blockDim.x;
  for (size_t i = (size_t)blockIdx.x * blockDim.x + threadIdx.x; i < total;
       i += step) {
    const size_t l  = i & (SEQL - 1);
    const size_t db = i >> 12;
    const size_t b  = db & (BATCH - 1);
    const size_t d  = db >> 4;
    uh[i] = (_Float16)u[((b * D_MODEL + d) << 12) + l];
  }
}

// ---------------------------------------------------------------------------
// K0: zero the pooled accumulator (workspace is poisoned by the harness)
// ---------------------------------------------------------------------------
__global__ void s4d_zero(float* __restrict__ p, int n) {
  int i = blockIdx.x * blockDim.x + threadIdx.x;
  if (i < n) p[i] = 0.0f;
}

// ---------------------------------------------------------------------------
// K2: causal depthwise conv as Toeplitz WMMA matmul, one block per channel d.
//   y(16 x 4096) = U(16 x 4096) x T_d  (lower-triangular Toeplitz of K[d])
// U panel staged LDS via the Tensor Data Mover (one tensor_load_to_lds per
// workgroup; waves fill the reversed-K tables concurrently).
// Epilogue: y += D[d]*u, exact GELU, store f16 into yact[(b*L+l)*512 + d].
// ---------------------------------------------------------------------------
#define KREV_N 4160                       // 4096 + 64 zero pad (t<0 region)
#define U_BYTES (BATCH * SEQL * 2)        // 131072
#define KREV_BYTES (KREV_N * 2)           // 8320
#define K2_LDS (U_BYTES + 2 * KREV_BYTES) // 147712

__global__ void s4d_conv_wmma(const _Float16* __restrict__ uh,
                              const float* __restrict__ Dvec,
                              const _Float16* __restrict__ Kf16,
                              _Float16* __restrict__ yact) {
  extern __shared__ char smem[];
  _Float16* Uh    = (_Float16*)smem;                         // [16][4096]
  _Float16* KrevE = (_Float16*)(smem + U_BYTES);             // Krev[i]
  _Float16* KrevO = (_Float16*)(smem + U_BYTES + KREV_BYTES);// Krev[i+1]

  const int d = blockIdx.x;
  const int tid = threadIdx.x;

  // --- Wave 0 kicks the TDM: DMA 128KB contiguous channel panel -> LDS. ---
  if (tid < 32) {
    const unsigned long long ga =
        (unsigned long long)(uh + (size_t)d * (BATCH * SEQL));
    // D# group 0: count=1 | lds_addr=0 | global_addr | type=2 (bits 127:126)
    v4u g0;
    g0[0] = 1u;                                   // count=1, user mode
    g0[1] = 0u;                                   // lds_addr = 0 (Uh base)
    g0[2] = (unsigned int)ga;                     // global_addr[31:0]
    g0[3] = ((unsigned int)(ga >> 32) & 0x01FFFFFFu) | 0x80000000u;
    // D# group 1: data_size=8B; tensor_dim0=tile_dim0=16384; 1 row.
    v8u g1;
    g1[0] = 0x00030000u;    // workgroup_mask=0, data_size=3 (8 bytes)
    g1[1] = 0x40000000u;    // tensor_dim0[15:0]=0x4000 in bits 63:48
    g1[2] = 0x00010000u;    // tensor_dim1=1 (bits 95:80)
    g1[3] = 0x40000000u;    // tile_dim0=0x4000 (bits 127:112)
    g1[4] = 0x00000001u;    // tile_dim1=1, tile_dim2=0
    g1[5] = 0x00004000u;    // tensor_dim0_stride[31:0]=16384
    g1[6] = 0u;
    g1[7] = 0u;
    asm volatile("tensor_load_to_lds %0, %1" :: "s"(g0), "s"(g1) : "memory");
  }

  // --- All waves: reversed kernel with zero pad (overlaps the DMA). ---
  // Krev[r] = K[d, 4095 - r], zero for r > 4095; KrevO[i] = Krev[i+1].
  const _Float16* Kd = Kf16 + d * SEQL;
  for (int r = tid; r < KREV_N; r += blockDim.x) {
    _Float16 ve = (r <= 4095) ? Kd[4095 - r] : (_Float16)0.0f;
    _Float16 vo = (r <= 4094) ? Kd[4094 - r] : (_Float16)0.0f;
    KrevE[r] = ve;
    KrevO[r] = vo;
  }
  if (tid < 32) __builtin_amdgcn_s_wait_tensorcnt(0);
  __syncthreads();

  const int wave = tid >> 5;
  const int lane = tid & 31;
  const int n    = lane & 15;   // A row (batch) / B col (l)
  const int hi   = lane >> 4;   // K-half selector
  const float Dd = Dvec[d];

  for (int T = wave; T < SEQL / 16; T += 8) {     // 8 waves per block
    const int l0 = T << 4;
    const int nch = (l0 + 16 + 31) >> 5;          // k-chunks: k0 in [0, l0]
    v8f acc = {};
    for (int c = 0; c < nch; ++c) {
      const int k0 = c << 5;
      // A tile: U[batch=n, k0 + hi*16 .. +15]  (two 16B LDS loads)
      union { uint4 q[2]; v16h v; } A;
      const char* ap = (const char*)Uh + (size_t)((n << 12) + k0 + (hi << 4)) * 2;
      A.q[0] = *(const uint4*)ap;
      A.q[1] = *(const uint4*)(ap + 16);
      // B tile (Toeplitz band): B[kk,n] = Krev[s0 + kk], contiguous per lane;
      // even/odd-shifted copies keep the dword loads 4B-aligned.
      const int s0 = 4095 - l0 - n + k0 + (hi << 4);
      const unsigned int* bp =
          (const unsigned int*)((s0 & 1) ? KrevO : KrevE) + (s0 >> 1);
      union { unsigned int w[8]; v16h v; } B;
      #pragma unroll
      for (int j = 0; j < 8; ++j) B.w[j] = bp[j];
      acc = __builtin_amdgcn_wmma_f32_16x16x32_f16(
          false, A.v, false, B.v, (short)0, acc, false, false);
    }
    // Epilogue: +D*u, exact GELU, store f16 in (col, channel) layout.
    #pragma unroll
    for (int r = 0; r < 8; ++r) {
      const int M = r + (hi << 3);                // batch row
      const int col = (M << 12) + l0 + n;         // b*L + l
      const float uval = (float)Uh[col];
      const float y = acc[r] + Dd * uval;
      const float g = 0.5f * y * (1.0f + erff(y * 0.70710678118654752f));
      yact[(size_t)col * D_MODEL + d] = (_Float16)g;
    }
  }
}

// ---------------------------------------------------------------------------
// K3: 1x1 conv (512 -> 1024) as WMMA GEMM + GLU + mean-pool accumulation.
// Block = (o_tile, b); each wave owns 32 l-tiles, computes paired a/g rows,
// applies a*sigmoid(g), reduces columns, atomically adds into pooled[b, o].
// ---------------------------------------------------------------------------
__global__ void s4d_glu_gemm(const _Float16* __restrict__ yact,
                             const float* __restrict__ W_out,
                             const float* __restrict__ b_out,
                             float* __restrict__ pooled) {
  __shared__ _Float16 Wa[16 * D_MODEL];   // rows o0..o0+15
  __shared__ _Float16 Wg[16 * D_MODEL];   // rows o0+512..
  __shared__ float ba[16], bg[16];

  const int ot = blockIdx.x >> 4;         // 0..31
  const int b  = blockIdx.x & 15;
  const int tid = threadIdx.x;

  for (int idx = tid; idx < 16 * D_MODEL; idx += blockDim.x) {
    const int row = idx >> 9, c = idx & (D_MODEL - 1);
    Wa[idx] = (_Float16)W_out[(size_t)(ot * 16 + row) * D_MODEL + c];
    Wg[idx] = (_Float16)W_out[(size_t)(ot * 16 + row + D_MODEL) * D_MODEL + c];
  }
  if (tid < 16)      ba[tid] = b_out[ot * 16 + tid];
  else if (tid < 32) bg[tid - 16] = b_out[ot * 16 + (tid - 16) + D_MODEL];
  __syncthreads();

  const int wave = tid >> 5;
  const int lane = tid & 31;
  const int n    = lane & 15;
  const int hi   = lane >> 4;

  v8f rows = {};
  const int Tend = wave * 32 + 32;
  for (int T = wave * 32; T < Tend; ++T) {
    const size_t col0 = ((size_t)b << 12) + ((size_t)T << 4);
    const char* bbase = (const char*)yact +
                        ((col0 + n) * D_MODEL + (hi << 4)) * 2;
    if (T + 1 < Tend)
      __builtin_prefetch(bbase + 16 * D_MODEL * 2, 0, 1); // next l-tile stream
    v8f aa = {}, ag = {};
    for (int kc = 0; kc < 16; ++kc) {
      const int c0 = kc << 5;
      union { uint4 q[2]; v16h v; } Aa, Ag, Bt;
      const char* pa = (const char*)Wa + (size_t)((n << 9) + c0 + (hi << 4)) * 2;
      const char* pg = (const char*)Wg + (size_t)((n << 9) + c0 + (hi << 4)) * 2;
      Aa.q[0] = *(const uint4*)pa;       Aa.q[1] = *(const uint4*)(pa + 16);
      Ag.q[0] = *(const uint4*)pg;       Ag.q[1] = *(const uint4*)(pg + 16);
      const char* pb = bbase + (size_t)c0 * 2;
      Bt.q[0] = *(const uint4*)pb;       Bt.q[1] = *(const uint4*)(pb + 16);
      aa = __builtin_amdgcn_wmma_f32_16x16x32_f16(
          false, Aa.v, false, Bt.v, (short)0, aa, false, false);
      ag = __builtin_amdgcn_wmma_f32_16x16x32_f16(
          false, Ag.v, false, Bt.v, (short)0, ag, false, false);
    }
    #pragma unroll
    for (int r = 0; r < 8; ++r) {
      const int M = r + (hi << 3);
      const float av = aa[r] + ba[M];
      const float gv = ag[r] + bg[M];
      rows[r] += av * (1.0f / (1.0f + expf(-gv)));   // GLU
    }
  }
  // Reduce the 16 columns held across the 16-lane groups of the D layout.
  #pragma unroll
  for (int r = 0; r < 8; ++r) {
    float v = rows[r];
    v += __shfl_xor(v, 1);
    v += __shfl_xor(v, 2);
    v += __shfl_xor(v, 4);
    v += __shfl_xor(v, 8);
    rows[r] = v;
  }
  if ((lane & 15) == 0) {
    #pragma unroll
    for (int r = 0; r < 8; ++r) {
      const int M = r + (hi << 3);
      atomicAdd(&pooled[b * D_MODEL + ot * 16 + M], rows[r]);
    }
  }
}

// ---------------------------------------------------------------------------
// K4: decoder — out[b] = mean_l(glu)[b,:] . W_dec + b_dec
// ---------------------------------------------------------------------------
__global__ void s4d_decode(const float* __restrict__ pooled,
                           const float* __restrict__ W_dec,
                           const float* __restrict__ b_dec,
                           float* __restrict__ out) {
  const int b = blockIdx.x;
  const int tid = threadIdx.x;            // 64 threads
  float s = 0.0f;
  for (int o = tid; o < D_MODEL; o += 64)
    s += pooled[b * D_MODEL + o] * W_dec[o];
  __shared__ float red[64];
  red[tid] = s;
  __syncthreads();
  for (int off = 32; off > 0; off >>= 1) {
    if (tid < off) red[tid] += red[tid + off];
    __syncthreads();
  }
  if (tid == 0) out[b] = red[0] * (1.0f / (float)SEQL) + b_dec[0];
}

// ---------------------------------------------------------------------------
extern "C" void kernel_launch(void* const* d_in, const int* in_sizes, int n_in,
                              void* d_out, int out_size, void* d_ws, size_t ws_size,
                              hipStream_t stream) {
  const float* u          = (const float*)d_in[0];
  const float* log_dt     = (const float*)d_in[1];
  const float* log_A_real = (const float*)d_in[2];
  const float* A_imag     = (const float*)d_in[3];
  const float* B_re       = (const float*)d_in[4];
  const float* B_im       = (const float*)d_in[5];
  const float* C_re       = (const float*)d_in[6];
  const float* C_im       = (const float*)d_in[7];
  const float* Dvec       = (const float*)d_in[8];
  const float* W_out      = (const float*)d_in[9];
  const float* b_out      = (const float*)d_in[10];
  const float* W_dec      = (const float*)d_in[11];
  const float* b_dec      = (const float*)d_in[12];

  char* ws = (char*)d_ws;
  _Float16* Kf16 = (_Float16*)ws;                                    // 4 MB
  _Float16* uh   = (_Float16*)(ws + ((size_t)4 << 20));              // 64 MB
  _Float16* yact = (_Float16*)(ws + ((size_t)68 << 20));             // 64 MB
  float*  pooled = (float*)(ws + ((size_t)132 << 20));               // 32 KB

  s4d_build_kernel<<<D_MODEL, 256, 0, stream>>>(
      log_dt, log_A_real, A_imag, B_re, B_im, C_re, C_im, Kf16);
  s4d_u_to_f16<<<8192, 256, 0, stream>>>(u, uh);
  s4d_zero<<<(BATCH * D_MODEL + 255) / 256, 256, 0, stream>>>(
      pooled, BATCH * D_MODEL);
  s4d_conv_wmma<<<D_MODEL, 256, K2_LDS, stream>>>(uh, Dvec, Kf16, yact);
  s4d_glu_gemm<<<32 * BATCH, 256, 0, stream>>>(yact, W_out, b_out, pooled);
  s4d_decode<<<BATCH, 64, 0, stream>>>(pooled, W_dec, b_dec, (float*)d_out);
}
